// WeightlessPhi3FFN_48241072668862
// MI455X (gfx1250) — matrix operand order repaired and checked by reading, not compiled, not run on hardware
//
#include <hip/hip_runtime.h>

// ---------------------------------------------------------------------------
// Phi3-style FFN, int8-dequant weights, for MI455X (gfx1250, wave32, WMMA).
//   hidden = silu(x@Wg^T * sg) * (x@Wu^T * su)      [fused, kernel 1]
//   out    = hidden@Wd^T * sd                        [kernel 2, TDM-staged A]
// f16 WMMA (weights 0..127 are exact in f16), f32 accumulation.
// ---------------------------------------------------------------------------

typedef __attribute__((ext_vector_type(16))) _Float16 v16h;
typedef __attribute__((ext_vector_type(8)))  _Float16 v8h;
typedef __attribute__((ext_vector_type(4)))  _Float16 v4h;
typedef __attribute__((ext_vector_type(8)))  float    v8f;
typedef __attribute__((ext_vector_type(4)))  unsigned int v4u;
typedef __attribute__((ext_vector_type(8)))  int      v8i;
typedef __attribute__((ext_vector_type(4)))  int      v4i;

#define HIDDEN 3072
#define INTER  8192
#define MTOK   8192   // B*S = 4*2048

#define BM 128        // token rows per block
#define BN 64         // output channels per block (per matrix)
#define BK 64         // K per LDS stage (2 WMMA k-steps)
#define LDK (BK + 8)  // padded LDS row stride in halves (144B = 36 DWORDs)

// Build a v16h fragment from two 16-byte LDS chunks.
__device__ __forceinline__ v16h ld_frag2(const _Float16* p0, const _Float16* p1) {
    v8h lo = *(const v8h*)p0;
    v8h hi = *(const v8h*)p1;
    return __builtin_shufflevector(lo, hi, 0,1,2,3,4,5,6,7,8,9,10,11,12,13,14,15);
}

// A-matrix fragment (16x32 f16) per ISA 7.12.2:
//   lanes 0-15  (half=0): row M=lane,    K = {kk+0..7,  kk+16..23}
//   lanes 16-31 (half=1): row M=lane-16, K = {kk+8..15, kk+24..31}
__device__ __forceinline__ v16h ld_afrag(const _Float16* sA, int mrow, int kk, int half) {
    const _Float16* base = sA + mrow * LDK + kk;
    return ld_frag2(base + 8 * half, base + 16 + 8 * half);
}

// B-matrix fragment (32x16 f16): lane holds column N=lane%16,
// K = kk + 8*(lane/16) + {0..7, 16..23}  (same K order as the A fragment).
__device__ __forceinline__ v16h ld_bfrag(const _Float16* sB, int ncol, int kk, int half) {
    const _Float16* base = sB + ncol * LDK + kk + 8 * half;
    return ld_frag2(base, base + 16);
}

// Stage a 64x64 int32 weight tile -> f16 LDS tile (exact: values in 0..127).
__device__ __forceinline__ void stage_w_tile(const int* __restrict__ w, int row0,
                                             int ld, int k0, _Float16* sB, int tid) {
#pragma unroll
    for (int it = 0; it < 4; ++it) {
        int id  = tid + 256 * it;        // 0..1023  (64 rows x 16 int4 chunks)
        int row = id >> 4;
        int c4  = (id & 15) << 2;
        const int4 wv = *(const int4*)(w + (size_t)(row0 + row) * ld + k0 + c4);
        v4h hv = { (_Float16)wv.x, (_Float16)wv.y, (_Float16)wv.z, (_Float16)wv.w };
        *(v4h*)(&sB[row * LDK + c4]) = hv;
    }
}

// ---------------------------------------------------------------------------
// TDM: async-DMA a [tile_rows x 64] f16 tile (row stride `ld_elems` in memory)
// into LDS at `lds_byte_off`, padding 4 DWORDs after every 32 data DWORDs so
// the LDS row stride is LDK halves (144 B). Issued by one wave; TENSORcnt.
// D# layout per CDNA5 ISA 8.3/8.4 (group0/group1; 2D tensor -> groups 2,3 = 0)
// ---------------------------------------------------------------------------
__device__ __forceinline__ void tdm_load_tile_f16(const _Float16* gsrc,
                                                  unsigned lds_byte_off,
                                                  unsigned tile_rows,
                                                  unsigned ld_elems) {
    unsigned long long ga = (unsigned long long)(const void*)gsrc;  // byte addr
    v4u g0;
    g0[0] = 1u;                                   // count=1, user mode, no gather
    g0[1] = lds_byte_off;                         // lds_addr
    g0[2] = (unsigned)(ga & 0xFFFFFFFFu);         // global_addr[31:0]
    g0[3] = (unsigned)((ga >> 32) & 0x01FFFFFFu)  // global_addr[56:32]
          | (2u << 30);                           // type=2 ("image")
    v8i g1;
    g1[0] = (int)((1u << 16)        // data_size = 1 (2 bytes)
          |       (1u << 20)        // pad_enable
          |       (4u << 22)        // pad_interval: 2^(4+1)=32 DWORDs per row
          |       (3u << 25));      // pad_amount: 3+1 = 4 DWORDs (16 B)
    g1[1] = (int)((ld_elems & 0xFFFFu) << 16);          // tensor_dim0[15:0]
    g1[2] = (int)(((ld_elems >> 16) & 0xFFFFu)          // tensor_dim0[31:16]
          |       ((unsigned)MTOK & 0xFFFFu) << 16);    // tensor_dim1[15:0]
    g1[3] = (int)((((unsigned)MTOK >> 16) & 0xFFFFu)    // tensor_dim1[31:16]
          |       (64u << 16));                         // tile_dim0 = 64 elems
    g1[4] = (int)(tile_rows & 0xFFFFu);                 // tile_dim1; tile_dim2=0
    g1[5] = (int)ld_elems;                              // tensor_dim0_stride lo
    g1[6] = 0;                                          // stride hi / dim1_stride
    g1[7] = 0;
    v4i gz4 = {0, 0, 0, 0};
    v8i gz8 = {0, 0, 0, 0, 0, 0, 0, 0};
    // 6-arg form (clang-23 / therock headers): (g0, g1, g2, g3, g4, cpol)
    __builtin_amdgcn_tensor_load_to_lds(g0, g1, gz4, gz4, gz8, 0);
}

// ---------------------------------------------------------------------------
// Kernel 1: up_states = x @ gate_up_w^T ; hidden = silu(gate*sg) * (up*su)
// ---------------------------------------------------------------------------
__global__ __launch_bounds__(256)
void ffn_gateup_silu_kernel(const float* __restrict__ x,        // [MTOK, HIDDEN]
                            const int*   __restrict__ w,        // [2*INTER, HIDDEN]
                            const float* __restrict__ scales,   // [2*INTER]
                            _Float16*    __restrict__ hidden) { // [MTOK, INTER]
    __shared__ _Float16 sA [BM * LDK];
    __shared__ _Float16 sBg[BN * LDK];
    __shared__ _Float16 sBu[BN * LDK];

    const int tid  = threadIdx.x;
    const int n0   = blockIdx.x * BN;   // hidden-dim column tile
    const int m0   = blockIdx.y * BM;   // token tile
    const int wave = tid >> 5;
    const int lane = tid & 31;
    const int wm   = wave & 3;          // 4 waves over M
    const int wn   = wave >> 2;         // 2 waves over N
    const int half = lane >> 4;
    const int l16  = lane & 15;

    v8f accG[2][2], accU[2][2];
#pragma unroll
    for (int i = 0; i < 2; ++i)
#pragma unroll
        for (int j = 0; j < 2; ++j) {
            v8f z = {0.f,0.f,0.f,0.f,0.f,0.f,0.f,0.f};
            accG[i][j] = z; accU[i][j] = z;
        }

    for (int k0 = 0; k0 < HIDDEN; k0 += BK) {
        __syncthreads();
        // Stage A: 128x64 fp32 -> f16 (each thread: 8 float4 loads)
#pragma unroll
        for (int it = 0; it < 8; ++it) {
            int id  = tid + 256 * it;    // 0..2047
            int row = id >> 4;
            int c4  = (id & 15) << 2;
            const float4 v = *(const float4*)(x + (size_t)(m0 + row) * HIDDEN + k0 + c4);
            v4h hv = { (_Float16)v.x, (_Float16)v.y, (_Float16)v.z, (_Float16)v.w };
            *(v4h*)(&sA[row * LDK + c4]) = hv;
        }
        // Stage B: gate rows [n0..) and up rows [INTER+n0..)
        stage_w_tile(w, n0,         HIDDEN, k0, sBg, tid);
        stage_w_tile(w, INTER + n0, HIDDEN, k0, sBu, tid);

        if (k0 + BK < HIDDEN) {       // advisory prefetch of next K tile
            __builtin_prefetch(x + (size_t)(m0 + (tid & 127)) * HIDDEN + k0 + BK, 0, 1);
            __builtin_prefetch(w + (size_t)(n0 + (tid & 63)) * HIDDEN + k0 + BK, 0, 1);
            __builtin_prefetch(w + (size_t)(INTER + n0 + (tid & 63)) * HIDDEN + k0 + BK, 0, 1);
        }
        __syncthreads();

#pragma unroll
        for (int kk = 0; kk < BK; kk += 32) {
            v16h afrag[2];
#pragma unroll
            for (int i = 0; i < 2; ++i)
                afrag[i] = ld_afrag(sA, 32 * wm + 16 * i + l16, kk, half);
#pragma unroll
            for (int j = 0; j < 2; ++j) {
                int ncol = 32 * wn + 16 * j + l16;
                v16h bG = ld_bfrag(sBg, ncol, kk, half);
                v16h bU = ld_bfrag(sBu, ncol, kk, half);
#pragma unroll
                for (int i = 0; i < 2; ++i) {
                    accG[i][j] = __builtin_amdgcn_wmma_f32_16x16x32_f16(
                        false, afrag[i], false, bG, (short)0, accG[i][j], false, false);
                    accU[i][j] = __builtin_amdgcn_wmma_f32_16x16x32_f16(
                        false, afrag[i], false, bU, (short)0, accU[i][j], false, false);
                }
            }
        }
    }

    // Epilogue: scale, SiLU*up, store f16 hidden.
    // D layout: lane=half*16+n holds rows M = r + 8*half (r=0..7), col N=lane%16.
#pragma unroll
    for (int j = 0; j < 2; ++j) {
        int ncol = n0 + 32 * wn + 16 * j + l16;
        float sg = scales[ncol];
        float su = scales[INTER + ncol];
#pragma unroll
        for (int i = 0; i < 2; ++i) {
            int mbase = m0 + 32 * wm + 16 * i + 8 * half;
#pragma unroll
            for (int r = 0; r < 8; ++r) {
                float g = accG[i][j][r] * sg;
                float u = accU[i][j][r] * su;
                float h = (g / (1.0f + __expf(-g))) * u;   // silu(g)*u
                hidden[(size_t)(mbase + r) * INTER + ncol] = (_Float16)h;
            }
        }
    }
}

// ---------------------------------------------------------------------------
// Kernel 2: out = hidden @ down_w^T * down_scales.  A-tile staged by TDM.
// ---------------------------------------------------------------------------
__global__ __launch_bounds__(256)
void ffn_down_kernel(const _Float16* __restrict__ hidden,  // [MTOK, INTER] f16
                     const int*      __restrict__ w,       // [HIDDEN, INTER]
                     const float*    __restrict__ scales,  // [HIDDEN]
                     float*          __restrict__ out) {   // [MTOK, HIDDEN]
    __shared__ _Float16 sA[BM * LDK];
    __shared__ _Float16 sB[BN * LDK];

    const int tid  = threadIdx.x;
    const int n0   = blockIdx.x * BN;
    const int m0   = blockIdx.y * BM;
    const int wave = tid >> 5;
    const int lane = tid & 31;
    const int wm   = wave & 3;
    const int wn   = wave >> 2;
    const int half = lane >> 4;
    const int l16  = lane & 15;

    const unsigned sA_off = (unsigned)(size_t)(&sA[0]);   // LDS byte offset

    v8f acc[2][2];
#pragma unroll
    for (int i = 0; i < 2; ++i)
#pragma unroll
        for (int j = 0; j < 2; ++j) {
            v8f z = {0.f,0.f,0.f,0.f,0.f,0.f,0.f,0.f};
            acc[i][j] = z;
        }

    for (int k0 = 0; k0 < INTER; k0 += BK) {
        __syncthreads();   // previous iteration finished reading LDS

        // A tile [BM x BK] f16: one wave kicks the Tensor Data Mover; the DMA
        // writes LDS with a 16B pad every 128B so rows land at stride LDK.
        if (wave == 0)
            tdm_load_tile_f16(hidden + (size_t)m0 * INTER + k0, sA_off, BM, INTER);

        // Meanwhile all waves stage + dequant the weight tile (int32 -> f16).
        stage_w_tile(w, n0, INTER, k0, sB, tid);

        if (k0 + BK < INTER)
            __builtin_prefetch(w + (size_t)(n0 + (tid & 63)) * INTER + k0 + BK, 0, 1);

        __builtin_amdgcn_s_wait_tensorcnt(0);   // no-op for non-issuing waves
        __syncthreads();

#pragma unroll
        for (int kk = 0; kk < BK; kk += 32) {
            v16h afrag[2];
#pragma unroll
            for (int i = 0; i < 2; ++i)
                afrag[i] = ld_afrag(sA, 32 * wm + 16 * i + l16, kk, half);
#pragma unroll
            for (int j = 0; j < 2; ++j) {
                v16h bf = ld_bfrag(sB, 32 * wn + 16 * j + l16, kk, half);
#pragma unroll
                for (int i = 0; i < 2; ++i)
                    acc[i][j] = __builtin_amdgcn_wmma_f32_16x16x32_f16(
                        false, afrag[i], false, bf, (short)0, acc[i][j], false, false);
            }
        }
    }

#pragma unroll
    for (int j = 0; j < 2; ++j) {
        int ncol = n0 + 32 * wn + 16 * j + l16;
        float sc = scales[ncol];
#pragma unroll
        for (int i = 0; i < 2; ++i) {
            int mbase = m0 + 32 * wm + 16 * i + 8 * half;
#pragma unroll
            for (int r = 0; r < 8; ++r)
                out[(size_t)(mbase + r) * HIDDEN + ncol] = acc[i][j][r] * sc;
        }
    }
}

// ---------------------------------------------------------------------------
extern "C" void kernel_launch(void* const* d_in, const int* in_sizes, int n_in,
                              void* d_out, int out_size, void* d_ws, size_t ws_size,
                              hipStream_t stream) {
    const float* x   = (const float*)d_in[0];
    const int*   guw = (const int*)d_in[1];
    const float* gus = (const float*)d_in[2];
    const int*   dw  = (const int*)d_in[3];
    const float* dsc = (const float*)d_in[4];
    float*       out = (float*)d_out;

    _Float16* hidden = (_Float16*)d_ws;   // [MTOK, INTER] f16 = 128 MB scratch

    dim3 g1(INTER / BN, MTOK / BM);   // 128 x 64 blocks
    ffn_gateup_silu_kernel<<<g1, 256, 0, stream>>>(x, guw, gus, hidden);

    dim3 g2(HIDDEN / BN, MTOK / BM); // 48 x 64 blocks
    ffn_down_kernel<<<g2, 256, 0, stream>>>(hidden, dw, dsc, out);
}
